// HierContextHead_42545946034820
// MI455X (gfx1250) — compile-verified, matrix operand order checked by reading, not run on hardware
//
#include <hip/hip_runtime.h>
#include <hip/hip_bf16.h>

// ---------------------------------------------------------------------------
// Problem constants (from reference)
// ---------------------------------------------------------------------------
constexpr int Bc  = 2;
constexpr int T   = 2048;
constexpr int E   = 1024;
constexpr int H   = 16;
constexpr int DH  = 64;          // E / H
constexpr int M1  = 1024;
constexpr int M2  = 256;
constexpr int BT  = Bc * T;      // 4096

typedef __attribute__((ext_vector_type(16))) __bf16        v16bf;
typedef __attribute__((ext_vector_type(8)))  __bf16        v8bf;
typedef __attribute__((ext_vector_type(8)))  float         v8f;
typedef __attribute__((ext_vector_type(4)))  unsigned int  v4u;
typedef __attribute__((ext_vector_type(8)))  int           v8i;
typedef __attribute__((ext_vector_type(4)))  int           v4i;

// ---------------------------------------------------------------------------
// WMMA helpers
// ---------------------------------------------------------------------------
__device__ __forceinline__ v8f wmma_bf16(v16bf a, v16bf b, v8f c) {
  return __builtin_amdgcn_wmma_f32_16x16x32_bf16(false, a, false, b,
                                                 (short)0, c, false, false);
}

// Load a 16x32 bf16 fragment (A layout; identical pattern serves as the B
// fragment for row-major "weightsᵀ"/"Kᵀ" operands).  Works for both global
// and LDS sources (addrspace inferred after inlining).  Per ISA 7.12.2:
// lanes 0-15 hold k in [k0,k0+8)∪[k0+16,k0+24); lanes 16-31 hold +8.
__device__ __forceinline__ v16bf frag_load(const __bf16* __restrict__ base,
                                           int ld, int row_base, int k0) {
  const int lane = threadIdx.x & 31;
  const int half = lane >> 4;
  const __bf16* p =
      base + (size_t)(row_base + (lane & 15)) * ld + k0 + half * 8;
  v8bf lo = *reinterpret_cast<const v8bf*>(p);        // k .. k+7
  v8bf hi = *reinterpret_cast<const v8bf*>(p + 16);   // k+16 .. k+23
  v16bf r;
#pragma unroll
  for (int i = 0; i < 8; ++i) { r[i] = lo[i]; r[8 + i] = hi[i]; }
  return r;
}

// ---------------------------------------------------------------------------
// Tensor Data Mover: one-shot 2-D panel load into LDS.
// D# built per CDNA5 ISA ch.8: group0 = {count/flags, lds_addr, global_addr,
// type=2}; group1 = {data_size=2B, tensor_dim0/1, tile_dim0/1, dim0 stride};
// groups 2/3 zero (<=2-D tensor).  Tracked with TENSORcnt.
// Toolchain uses the 6-arg builtin: (v4u, v8i, v4i, v4i, v8i, i32 cpol).
// ---------------------------------------------------------------------------
__device__ __forceinline__ void tdm_load_2d(unsigned lds_byte_off,
                                            const void* gptr,
                                            unsigned tensor_d0,
                                            unsigned tensor_d1,
                                            unsigned tile_d0, unsigned tile_d1,
                                            unsigned long stride0_elems) {
  const unsigned long ga = (unsigned long)(uintptr_t)gptr;
  v4u g0;
  g0[0] = 1u;                                         // count=1, user D#
  g0[1] = lds_byte_off;                               // lds_addr
  g0[2] = (unsigned)(ga & 0xFFFFFFFFu);               // global_addr[31:0]
  g0[3] = (unsigned)((ga >> 32) & 0x01FFFFFFu)        // global_addr[56:32]
          | (2u << 30);                               // type = 2 ("image")
  v8i g1;
  g1[0] = (int)(1u << 16);                            // data_size = 2 bytes
  g1[1] = (int)((tensor_d0 & 0xFFFFu) << 16);         // tensor_dim0 lo16
  g1[2] = (int)((tensor_d0 >> 16) | ((tensor_d1 & 0xFFFFu) << 16));
  g1[3] = (int)((tensor_d1 >> 16) | ((tile_d0 & 0xFFFFu) << 16));
  g1[4] = (int)(tile_d1 & 0xFFFFu);                   // tile_dim2 = 0
  g1[5] = (int)(stride0_elems & 0xFFFFFFFFul);        // tensor_dim0_stride
  g1[6] = (int)((stride0_elems >> 32) & 0xFFFFul);
  g1[7] = 0;
  v4i z4 = {};
  v8i z8 = {};
  __builtin_amdgcn_tensor_load_to_lds(g0, g1, z4, z4, z8, 0);
}

// ---------------------------------------------------------------------------
// 1) fp32 -> bf16 cast (grid-stride)
// ---------------------------------------------------------------------------
__global__ __launch_bounds__(256) void cast_bf16_kernel(
    const float* __restrict__ src, __bf16* __restrict__ dst, int n) {
  for (int i = blockIdx.x * blockDim.x + threadIdx.x; i < n;
       i += gridDim.x * blockDim.x)
    dst[i] = (__bf16)src[i];
}

// Vt[b][h][d][m] = V[b][m][h*DH + d]   (per-head transpose, bf16)
__global__ __launch_bounds__(256) void vtrans_kernel(
    const float* __restrict__ V, __bf16* __restrict__ Vt, int M) {
  const long n = (long)Bc * H * DH * M;
  for (long i = blockIdx.x * (long)blockDim.x + threadIdx.x; i < n;
       i += (long)gridDim.x * blockDim.x) {
    int  m = (int)(i % M);
    long r = i / M;
    int  d = (int)(r % DH); r /= DH;
    int  h = (int)(r % H);
    int  b = (int)(r / H);
    Vt[i] = (__bf16)V[((long)b * M + m) * E + h * DH + d];
  }
}

// ---------------------------------------------------------------------------
// 2/4) GEMM:  C[r][n] = sum_k A[r][k] * W[n][k]   (i.e. A @ Wᵀ)
//     One wave computes a 32x64 tile: 2 A-frags x 4 B-frags per k-step.
// ---------------------------------------------------------------------------
__device__ __forceinline__ void gemm_tile(const __bf16* __restrict__ A, int ldA,
                                          const __bf16* __restrict__ W, int ldW,
                                          int m0, int n0, int K,
                                          v8f acc[2][4]) {
  for (int k0 = 0; k0 < K; k0 += 32) {
    v16bf a0 = frag_load(A, ldA, m0,      k0);
    v16bf a1 = frag_load(A, ldA, m0 + 16, k0);
#pragma unroll
    for (int nt = 0; nt < 4; ++nt) {
      v16bf bt = frag_load(W, ldW, n0 + nt * 16, k0);
      acc[0][nt] = wmma_bf16(a0, bt, acc[0][nt]);
      acc[1][nt] = wmma_bf16(a1, bt, acc[1][nt]);
    }
  }
}

__global__ __launch_bounds__(256) void qgemm_kernel(
    const __bf16* __restrict__ X, const __bf16* __restrict__ Wq,
    __bf16* __restrict__ Q) {
  const int lane = threadIdx.x & 31;
  const int wave = threadIdx.x >> 5;
  const int tiles_n = E / 64;                       // 16
  const int tile = blockIdx.x * 8 + wave;
  if (tile >= (BT / 32) * tiles_n) return;
  const int m0 = (tile / tiles_n) * 32;
  const int n0 = (tile % tiles_n) * 64;

  v8f acc[2][4] = {};
  gemm_tile(X, E, Wq, E, m0, n0, E, acc);

  const int half = lane >> 4, nn = lane & 15;
#pragma unroll
  for (int mi = 0; mi < 2; ++mi)
#pragma unroll
    for (int nt = 0; nt < 4; ++nt)
#pragma unroll
      for (int i = 0; i < 8; ++i)
        Q[(size_t)(m0 + mi * 16 + half * 8 + i) * E + n0 + nt * 16 + nn] =
            (__bf16)acc[mi][nt][i];
}

__global__ __launch_bounds__(256) void projgemm_kernel(
    const __bf16* __restrict__ Ctx, const __bf16* __restrict__ Wp,
    const float* __restrict__ gate, float* __restrict__ Out) {
  const int lane = threadIdx.x & 31;
  const int wave = threadIdx.x >> 5;
  const int tiles_n = E / 64;
  const int tile = blockIdx.x * 8 + wave;
  if (tile >= (BT / 32) * tiles_n) return;
  const int m0 = (tile / tiles_n) * 32;
  const int n0 = (tile % tiles_n) * 64;

  const float g = gate[0];
  const float alpha = 1.0f / (1.0f + __expf(-g));   // sigmoid(gate)

  v8f acc[2][4] = {};
  gemm_tile(Ctx, E, Wp, E, m0, n0, E, acc);

  const int half = lane >> 4, nn = lane & 15;
#pragma unroll
  for (int mi = 0; mi < 2; ++mi)
#pragma unroll
    for (int nt = 0; nt < 4; ++nt)
#pragma unroll
      for (int i = 0; i < 8; ++i)
        Out[(size_t)(m0 + mi * 16 + half * 8 + i) * E + n0 + nt * 16 + nn] =
            alpha * acc[mi][nt][i];
}

// ---------------------------------------------------------------------------
// 3) Flash attention over one LDS-resident bank.  One wave owns 16 query rows;
//    the workgroup's 8 waves share the (b,h) K/V panels staged in LDS by the
//    Tensor Data Mover.  Scores in D-layout; row stats reduced with 4
//    xor-shuffles per 16-lane group; P re-laid-out D->A through a private
//    LDS tile.
// ---------------------------------------------------------------------------
__device__ __forceinline__ void attend_bank(
    const __bf16* __restrict__ kpan,   // LDS [M][DH], ld = DH
    const __bf16* __restrict__ vpan,   // LDS [DH][M], ld = M
    int M, v16bf aq0, v16bf aq1,
    float* __restrict__ P,             // 16x32 fp32 LDS tile (per wave)
    float total[4][8]) {
  const float scale = 0.125f;          // 1/sqrt(64)
  const int lane = threadIdx.x & 31;
  const int half = lane >> 4, nn = lane & 15;

  v8f acc[4] = {};
  float mrun[8], lrun[8];
#pragma unroll
  for (int i = 0; i < 8; ++i) { mrun[i] = -INFINITY; lrun[i] = 0.0f; }

  for (int j = 0; j < M; j += 32) {
    // ---- scores: two 16x16 tiles covering memory cols [j, j+32)
    v8f s0 = {}, s1 = {};
    {
      v16bf b00 = frag_load(kpan, DH, j,      0);
      v16bf b01 = frag_load(kpan, DH, j,      32);
      s0 = wmma_bf16(aq0, b00, s0);
      s0 = wmma_bf16(aq1, b01, s0);
      v16bf b10 = frag_load(kpan, DH, j + 16, 0);
      v16bf b11 = frag_load(kpan, DH, j + 16, 32);
      s1 = wmma_bf16(aq0, b10, s1);
      s1 = wmma_bf16(aq1, b11, s1);
    }

    // ---- online softmax update (per row i, owned by 16-lane group)
    float p0[8], p1[8];
#pragma unroll
    for (int i = 0; i < 8; ++i) {
      float a = s0[i] * scale, c = s1[i] * scale;
      float v = fmaxf(a, c);
#pragma unroll
      for (int msk = 1; msk < 16; msk <<= 1)
        v = fmaxf(v, __shfl_xor(v, msk, 32));
      const float mnew = fmaxf(mrun[i], v);
      const float e0 = __expf(a - mnew);
      const float e1 = __expf(c - mnew);
      p0[i] = e0; p1[i] = e1;
      float s = e0 + e1;
#pragma unroll
      for (int msk = 1; msk < 16; msk <<= 1)
        s += __shfl_xor(s, msk, 32);
      const float corr = __expf(mrun[i] - mnew);
      lrun[i] = lrun[i] * corr + s;
      mrun[i] = mnew;
#pragma unroll
      for (int nt = 0; nt < 4; ++nt) acc[nt][i] *= corr;
    }

    // ---- D-layout -> A-layout transpose of P through LDS
#pragma unroll
    for (int i = 0; i < 8; ++i) {
      const int row = half * 8 + i;
      P[row * 32 + nn]      = p0[i];
      P[row * 32 + 16 + nn] = p1[i];
    }
    asm volatile("s_wait_dscnt 0x0" ::: "memory");  // cross-lane RAW via LDS
    v16bf pa;
    {
      const int kb = half * 8;
#pragma unroll
      for (int i = 0; i < 8; ++i) {
        pa[i]     = (__bf16)P[nn * 32 + kb + i];
        pa[8 + i] = (__bf16)P[nn * 32 + 16 + kb + i];
      }
    }
    asm volatile("" ::: "memory");                  // keep loads before next stores

    // ---- ctx += P @ V   (B fragment = contiguous rows of LDS Vᵀ panel)
#pragma unroll
    for (int nt = 0; nt < 4; ++nt) {
      v16bf bv = frag_load(vpan, M, nt * 16, j);
      acc[nt] = wmma_bf16(pa, bv, acc[nt]);
    }
  }

  // ---- normalize and accumulate into the per-wave total
#pragma unroll
  for (int i = 0; i < 8; ++i) {
    const float inv = 1.0f / lrun[i];
#pragma unroll
    for (int nt = 0; nt < 4; ++nt) total[nt][i] += acc[nt][i] * inv;
  }
}

// Dynamic LDS layout (CDNA5: up to 320KB/WGP):
//   [0, 128KB)            K panel  [M1][DH] bf16
//   [128KB, 256KB)        Vᵀ panel [DH][M1] bf16
//   [256KB, 272KB)        8 x 16x32 fp32 P tiles
constexpr unsigned KPAN_ELEMS = (unsigned)M1 * DH;          // 65536
constexpr unsigned ATTN_LDS_BYTES = 2u * KPAN_ELEMS * 2u + 8u * 512u * 4u;

__global__ __launch_bounds__(256) void attn_kernel(
    const __bf16* __restrict__ Qbf,
    const __bf16* __restrict__ K1b, const __bf16* __restrict__ V1t,
    const __bf16* __restrict__ K2b, const __bf16* __restrict__ V2t,
    __bf16* __restrict__ Ctx) {
  extern __shared__ __align__(16) __bf16 dynlds[];
  __bf16* kpan = dynlds;                      // [M][DH]
  __bf16* vpan = dynlds + KPAN_ELEMS;         // [DH][M]
  float*  Pf   = (float*)(dynlds + 2 * KPAN_ELEMS);

  const int lane = threadIdx.x & 31;
  const int wave = threadIdx.x >> 5;
  const int half = lane >> 4, nn = lane & 15;

  const int TB = T / 128;                     // 16 t-blocks per (b,h)
  const int tb = blockIdx.x % TB;
  const int bh = blockIdx.x / TB;
  const int h = bh % H, b = bh / H;
  const int t0 = tb * 128 + wave * 16;        // 16 query rows per wave

  const __bf16* Qb = Qbf + (size_t)b * T * E;
  const v16bf aq0 = frag_load(Qb, E, t0, h * DH);
  const v16bf aq1 = frag_load(Qb, E, t0, h * DH + 32);

  float* myP = Pf + wave * 512;
  float total[4][8] = {};

  // ---- bank 1: TDM-stage the whole (b,h) K and Vᵀ panels into LDS
  if (threadIdx.x == 0) {
    tdm_load_2d((unsigned)(uintptr_t)kpan,
                K1b + (size_t)b * M1 * E + h * DH,
                DH, M1, DH, M1, (unsigned long)E);
    tdm_load_2d((unsigned)(uintptr_t)vpan,
                V1t + ((size_t)b * H + h) * DH * M1,
                M1, DH, M1, DH, (unsigned long)M1);
    __builtin_amdgcn_s_wait_tensorcnt(0);
  }
  __syncthreads();
  attend_bank(kpan, vpan, M1, aq0, aq1, myP, total);
  __syncthreads();                            // panels free for reuse

  // ---- bank 2: reuse the same LDS panels (32KB + 32KB used)
  if (threadIdx.x == 0) {
    tdm_load_2d((unsigned)(uintptr_t)kpan,
                K2b + (size_t)b * M2 * E + h * DH,
                DH, M2, DH, M2, (unsigned long)E);
    tdm_load_2d((unsigned)(uintptr_t)vpan,
                V2t + ((size_t)b * H + h) * DH * M2,
                M2, DH, M2, DH, (unsigned long)M2);
    __builtin_amdgcn_s_wait_tensorcnt(0);
  }
  __syncthreads();
  attend_bank(kpan, vpan, M2, aq0, aq1, myP, total);

#pragma unroll
  for (int nt = 0; nt < 4; ++nt)
#pragma unroll
    for (int i = 0; i < 8; ++i)
      Ctx[((size_t)b * T + t0 + half * 8 + i) * E + h * DH + nt * 16 + nn] =
          (__bf16)total[nt][i];
}

// ---------------------------------------------------------------------------
// Host launcher
// ---------------------------------------------------------------------------
extern "C" void kernel_launch(void* const* d_in, const int* in_sizes, int n_in,
                              void* d_out, int out_size, void* d_ws,
                              size_t ws_size, hipStream_t stream) {
  const float* x     = (const float*)d_in[0];
  const float* K1f   = (const float*)d_in[1];
  const float* V1f   = (const float*)d_in[2];
  const float* K2f   = (const float*)d_in[3];
  const float* V2f   = (const float*)d_in[4];
  const float* Wq    = (const float*)d_in[5];
  const float* Wproj = (const float*)d_in[6];
  const float* gate  = (const float*)d_in[7];
  float* out = (float*)d_out;

  // Workspace layout (bf16 elements)
  __bf16* w = (__bf16*)d_ws;
  __bf16* xbf   = w;                   w += (size_t)BT * E;        // 4M
  __bf16* qbf   = w;                   w += (size_t)BT * E;        // 4M
  __bf16* ctxbf = w;                   w += (size_t)BT * E;        // 4M
  __bf16* wqbf  = w;                   w += (size_t)E * E;         // 1M
  __bf16* wpbf  = w;                   w += (size_t)E * E;         // 1M
  __bf16* k1bf  = w;                   w += (size_t)Bc * M1 * E;   // 2M
  __bf16* k2bf  = w;                   w += (size_t)Bc * M2 * E;   // 0.5M
  __bf16* v1t   = w;                   w += (size_t)Bc * H * DH * M1;
  __bf16* v2t   = w;                   w += (size_t)Bc * H * DH * M2;
  (void)ws_size; (void)n_in; (void)in_sizes; (void)out_size;

  const int thr = 256;
  auto blocks = [](long n) { long b = (n + 255) / 256; return (int)(b > 4096 ? 4096 : b); };

  // 1) precision conversion + V transpose
  cast_bf16_kernel<<<blocks((long)BT * E), thr, 0, stream>>>(x, xbf, BT * E);
  cast_bf16_kernel<<<blocks((long)E * E), thr, 0, stream>>>(Wq, wqbf, E * E);
  cast_bf16_kernel<<<blocks((long)E * E), thr, 0, stream>>>(Wproj, wpbf, E * E);
  cast_bf16_kernel<<<blocks((long)Bc * M1 * E), thr, 0, stream>>>(K1f, k1bf, Bc * M1 * E);
  cast_bf16_kernel<<<blocks((long)Bc * M2 * E), thr, 0, stream>>>(K2f, k2bf, Bc * M2 * E);
  vtrans_kernel<<<blocks((long)Bc * H * DH * M1), thr, 0, stream>>>(V1f, v1t, M1);
  vtrans_kernel<<<blocks((long)Bc * H * DH * M2), thr, 0, stream>>>(V2f, v2t, M2);

  // 2) Q = x @ Wqᵀ  : (BT/32)*(E/64) = 2048 wave-tiles, 8 waves/block
  qgemm_kernel<<<(BT / 32) * (E / 64) / 8, thr, 0, stream>>>(xbf, wqbf, qbf);

  // 3) dual-bank flash attention; K/V panels TDM-staged into 272KB of LDS,
  //    shared by the 8 waves (128 query rows) of each (b,h) workgroup.
  attn_kernel<<<Bc * H * (T / 128), thr, ATTN_LDS_BYTES, stream>>>(
      qbf, k1bf, v1t, k2bf, v2t, ctxbf);

  // 4) out = sigmoid(gate) * (ctx @ Wprojᵀ)
  projgemm_kernel<<<(BT / 32) * (E / 64) / 8, thr, 0, stream>>>(ctxbf, wpbf,
                                                                gate, out);
}
